// LSTMAutoEncoder_2448131358987
// MI455X (gfx1250) — matrix operand order repaired and checked by reading, not compiled, not run on hardware
//
#include <hip/hip_runtime.h>

#define DIN   128
#define HID   64
#define DOUT  128
#define SEQT  1024
#define GATES 256   // 4*HID
#define BROWS 16    // batch rows per block
#define NBLK  16    // 16*16 = 256 = B
#define NTHR  512   // 16 waves

typedef __attribute__((ext_vector_type(16))) __bf16 v16bf;
typedef __attribute__((ext_vector_type(8)))  float  v8f;

union Frag { unsigned u[8]; v16bf v; };

// ---- f32 -> bf16 (round-to-nearest-even-ish) ----
__device__ __forceinline__ unsigned bf1(float a) {
  unsigned ua = __float_as_uint(a);
  return (ua + 0x7FFFu + ((ua >> 16) & 1u)) >> 16;
}
__device__ __forceinline__ unsigned pk(float a, float b) {
  return bf1(a) | (bf1(b) << 16);
}
__device__ __forceinline__ float bf2f(unsigned h) { return __uint_as_float(h << 16); }

// ---- fast sigmoid / tanh: v_exp_f32 + v_rcp_f32 ----
__device__ __forceinline__ float sg(float x) {
  return __builtin_amdgcn_rcpf(1.0f + __builtin_amdgcn_exp2f(-1.44269504f * x));
}
__device__ __forceinline__ float tnh(float x) { return 2.0f * sg(2.0f * x) - 1.0f; }

__device__ __forceinline__ v8f wmma_bf16(const Frag& a, const Frag& b, v8f c) {
  return __builtin_amdgcn_wmma_f32_16x16x32_bf16(false, a.v, false, b.v, (short)0, c,
                                                 false, false);
}

// B-fragment (weights) from global f32 row-major W[N][K].
// gates = act @ W^T  =>  B(k,n) = W[n][k]. Tile = 16 N-rows, chunk = 32 K.
// Assumed B layout: lane -> n = lane&15; K = (lane>>4)*8 + 2j (+1) for VGPR j<4; j>=4: +8 (K = half*8 + 16 + 2(j-4)).
__device__ __forceinline__ Frag loadB(const float* __restrict__ W, int K, int ntile,
                                      int chunk, int lane) {
  int n  = ntile * 16 + (lane & 15);
  int kb = chunk * 32 + (lane >> 4) * 8;
  const float* p = W + (size_t)n * K + kb;
  Frag f;
#pragma unroll
  for (int j = 0; j < 8; ++j) f.u[j] = pk(p[2 * j + (j >> 2) * 8], p[2 * j + 1 + (j >> 2) * 8]);
  return f;
}

// A-fragment from LDS bf16 row-major [16][stride]; ISA A layout:
// lane: m = lane&15, half = lane>>4; VGPR j<4: K = half*8 + 2j; j>=4: +16.
// => two 16B contiguous runs per lane: ds_load_b128 x2.
__device__ __forceinline__ Frag loadA(const unsigned short* base, int stride, int chunk,
                                      int lane) {
  int m = lane & 15, half = lane >> 4;
  const unsigned short* p = base + m * stride + chunk * 32 + half * 8;
  uint4 lo = *(const uint4*)(p);
  uint4 hi = *(const uint4*)(p + 16);
  Frag f;
  f.u[0] = lo.x; f.u[1] = lo.y; f.u[2] = lo.z; f.u[3] = lo.w;
  f.u[4] = hi.x; f.u[5] = hi.y; f.u[6] = hi.z; f.u[7] = hi.w;
  return f;
}

// One layer's gate GEMM for this wave's 16 gate columns: acc = bias + x@Wx^T + h@Wh^T
// All A-fragments are preloaded first so the ds_load_b128s issue back-to-back and
// LDS latency overlaps the WMMA accumulate chain (partial dscnt waits).
template <int NXC>
__device__ __forceinline__ void lstm_gates(const unsigned short* xbuf, int xstride,
                                           const unsigned short* hbuf, const Frag* Wx,
                                           const Frag* Wh, float bias, float* gbuf,
                                           int lane, int w) {
  Frag a[NXC + 2];
#pragma unroll
  for (int c = 0; c < NXC; ++c) a[c] = loadA(xbuf, xstride, c, lane);
#pragma unroll
  for (int c = 0; c < 2; ++c) a[NXC + c] = loadA(hbuf, HID, c, lane);
  v8f acc = {bias, bias, bias, bias, bias, bias, bias, bias};
#pragma unroll
  for (int c = 0; c < NXC; ++c) acc = wmma_bf16(a[c], Wx[c], acc);
#pragma unroll
  for (int c = 0; c < 2; ++c) acc = wmma_bf16(a[NXC + c], Wh[c], acc);
  int half = lane >> 4, col = w * 16 + (lane & 15);
#pragma unroll
  for (int v = 0; v < 8; ++v) gbuf[(v + 8 * half) * GATES + col] = acc[v];  // M=v+8*half
}

// Elementwise LSTM cell update; each thread owns one (row, hidden-pair).
__device__ __forceinline__ void cell_update(const float* gbuf, float* cb, unsigned* hw,
                                            int tid) {
  int jp = tid & 31, j = jp * 2;
  int m  = tid >> 5;
  const float* gr = gbuf + m * GATES;
  float i0 = sg(gr[j]),        i1 = sg(gr[j + 1]);
  float f0 = sg(gr[64 + j]),   f1 = sg(gr[64 + j + 1]);
  float g0 = tnh(gr[128 + j]), g1 = tnh(gr[128 + j + 1]);
  float o0 = sg(gr[192 + j]),  o1 = sg(gr[192 + j + 1]);
  float c0 = f0 * cb[m * HID + j] + i0 * g0;
  float c1 = f1 * cb[m * HID + j + 1] + i1 * g1;
  cb[m * HID + j]     = c0;
  cb[m * HID + j + 1] = c1;
  hw[m * (HID / 2) + jp] = pk(o0 * tnh(c0), o1 * tnh(c1));
}

__global__ __launch_bounds__(NTHR) void lstm_ae_cdna5(
    const float* __restrict__ x,
    const float* eW0x, const float* eW0h, const float* eb0i, const float* eb0h,
    const float* eW1x, const float* eW1h, const float* eb1i, const float* eb1h,
    const float* dW0x, const float* dW0h, const float* db0i, const float* db0h,
    const float* dW1x, const float* dW1h, const float* db1i, const float* db1h,
    const float* fcW, const float* fcBg, float* __restrict__ out) {
  __shared__ float gbuf[BROWS * GATES];                                    // 16 KB
  __shared__ float c0b[BROWS * HID], c1b[BROWS * HID];                     // 4+4 KB
  __shared__ __attribute__((aligned(16))) unsigned short xb0[BROWS * DIN]; // 4 KB bf16
  __shared__ __attribute__((aligned(16))) unsigned short xb1[BROWS * DIN]; // 4 KB bf16
  __shared__ __attribute__((aligned(16))) unsigned short h0b[BROWS * HID];
  __shared__ __attribute__((aligned(16))) unsigned short h1b[BROWS * HID];

  const int tid = threadIdx.x, lane = tid & 31, w = tid >> 5;
  const int r0 = blockIdx.x * BROWS;
  const int n  = lane & 15;

  // ---- zero initial state ----
  ((unsigned*)h0b)[tid] = 0u;   // BROWS*HID/2 == NTHR
  ((unsigned*)h1b)[tid] = 0u;
#pragma unroll
  for (int i = tid; i < BROWS * HID; i += NTHR) { c0b[i] = 0.0f; c1b[i] = 0.0f; }

  // ---- encoder weights into VGPR B-fragments (once) ----
  float b0 = eb0i[w * 16 + n] + eb0h[w * 16 + n];
  float b1 = eb1i[w * 16 + n] + eb1h[w * 16 + n];
  Frag Wx0[4], Wh0[2], Wx1[2], Wh1[2];
#pragma unroll
  for (int c = 0; c < 4; ++c) Wx0[c] = loadB(eW0x, DIN, w, c, lane);
#pragma unroll
  for (int c = 0; c < 2; ++c) {
    Wh0[c] = loadB(eW0h, HID, w, c, lane);
    Wx1[c] = loadB(eW1x, HID, w, c, lane);
    Wh1[c] = loadB(eW1h, HID, w, c, lane);
  }

  // ---- prologue: stage x_0 into bank 0 ----
  {
    int mm = tid >> 5, cq = tid & 31;
    const float* p = x + (size_t)(r0 + mm) * SEQT * DIN + cq * 4;
    unsigned* xw = (unsigned*)xb0;
    xw[mm * (DIN / 2) + cq * 2]     = pk(p[0], p[1]);
    xw[mm * (DIN / 2) + cq * 2 + 1] = pk(p[2], p[3]);
  }
  __syncthreads();

  // =================== encoder: 1024 fused 2-layer steps ===================
  // x is double-buffered: loads for x_{t+1} issue at the top of step t (VMEM
  // latency hidden under 10 WMMAs + 2 cell updates), stored to the idle bank at
  // the end, covered by the end-of-step barrier.  4 barriers/step.
  for (int t = 0; t < SEQT; ++t) {
    const unsigned short* xcur = (t & 1) ? xb1 : xb0;
    unsigned short*       xnxt = (t & 1) ? xb0 : xb1;
    const int mm = tid >> 5, cq = tid & 31;
    float nx0 = 0.f, nx1 = 0.f, nx2 = 0.f, nx3 = 0.f;
    if (t + 1 < SEQT) {  // early-issue next-step global loads
      const float* p =
          x + (size_t)(r0 + mm) * SEQT * DIN + (size_t)(t + 1) * DIN + cq * 4;
      nx0 = p[0]; nx1 = p[1]; nx2 = p[2]; nx3 = p[3];
      __builtin_prefetch(p + DIN, 0, 0);  // global_prefetch_b8 for t+2
    }
    lstm_gates<4>(xcur, DIN, h0b, Wx0, Wh0, b0, gbuf, lane, w);
    __syncthreads();
    cell_update(gbuf, c0b, (unsigned*)h0b, tid);
    __syncthreads();
    lstm_gates<2>(h0b, HID, h1b, Wx1, Wh1, b1, gbuf, lane, w);
    __syncthreads();
    cell_update(gbuf, c1b, (unsigned*)h1b, tid);
    if (t + 1 < SEQT) {
      unsigned* xw = (unsigned*)xnxt;
      xw[mm * (DIN / 2) + cq * 2]     = pk(nx0, nx1);
      xw[mm * (DIN / 2) + cq * 2 + 1] = pk(nx2, nx3);
    }
    __syncthreads();
  }

  // ---- tanh on carried states; zero decoder x0 (bank 0) ----
  {
    unsigned* hw0 = (unsigned*)h0b;
    unsigned* hw1 = (unsigned*)h1b;
    unsigned p0 = hw0[tid]; hw0[tid] = pk(tnh(bf2f(p0 & 0xFFFFu)), tnh(bf2f(p0 >> 16)));
    unsigned p1 = hw1[tid]; hw1[tid] = pk(tnh(bf2f(p1 & 0xFFFFu)), tnh(bf2f(p1 >> 16)));
#pragma unroll
    for (int i = tid; i < BROWS * HID; i += NTHR) { c0b[i] = tnh(c0b[i]); c1b[i] = tnh(c1b[i]); }
#pragma unroll
    for (int i = tid; i < BROWS * DIN / 2; i += NTHR) ((unsigned*)xb0)[i] = 0u;
  }

  // ---- decoder weights (reuse the same registers) ----
  b0 = db0i[w * 16 + n] + db0h[w * 16 + n];
  b1 = db1i[w * 16 + n] + db1h[w * 16 + n];
#pragma unroll
  for (int c = 0; c < 4; ++c) Wx0[c] = loadB(dW0x, DIN, w, c, lane);
#pragma unroll
  for (int c = 0; c < 2; ++c) {
    Wh0[c] = loadB(dW0h, HID, w, c, lane);
    Wx1[c] = loadB(dW1x, HID, w, c, lane);
    Wh1[c] = loadB(dW1h, HID, w, c, lane);
  }
  Frag Wfc[2] = {};
  float bf = 0.0f;
  if (w < 8) {  // 8 waves cover the 128 fc output columns
#pragma unroll
    for (int c = 0; c < 2; ++c) Wfc[c] = loadB(fcW, HID, w, c, lane);
    bf = fcBg[w * 16 + n];
  }
  __syncthreads();

  // =================== decoder: 1024 autoregressive steps ===================
  for (int t = 0; t < SEQT; ++t) {
    lstm_gates<4>(xb0, DIN, h0b, Wx0, Wh0, b0, gbuf, lane, w);
    __syncthreads();
    cell_update(gbuf, c0b, (unsigned*)h0b, tid);
    __syncthreads();
    lstm_gates<2>(h0b, HID, h1b, Wx1, Wh1, b1, gbuf, lane, w);
    __syncthreads();
    cell_update(gbuf, c1b, (unsigned*)h1b, tid);
    __syncthreads();
    if (w < 8) {  // fc: pred = h1 @ fcW^T + fcB ; write out + feed back (bf16)
      v8f P = {bf, bf, bf, bf, bf, bf, bf, bf};
#pragma unroll
      for (int c = 0; c < 2; ++c) P = wmma_bf16(loadA(h1b, HID, c, lane), Wfc[c], P);
      int half = lane >> 4, col = w * 16 + n;
#pragma unroll
      for (int v = 0; v < 8; ++v) {
        int m = v + 8 * half;
        out[(size_t)(r0 + m) * SEQT * DOUT + (size_t)t * DOUT + col] = P[v];
        xb0[m * DIN + col] = (unsigned short)bf1(P[v]);  // ds_store_b16
      }
    }
    __syncthreads();
  }
}

extern "C" void kernel_launch(void* const* d_in, const int* in_sizes, int n_in,
                              void* d_out, int out_size, void* d_ws, size_t ws_size,
                              hipStream_t stream) {
  (void)in_sizes; (void)n_in; (void)out_size; (void)d_ws; (void)ws_size;
  const float* p[19];
  for (int i = 0; i < 19; ++i) p[i] = (const float*)d_in[i];
  lstm_ae_cdna5<<<dim3(NBLK), dim3(NTHR), 0, stream>>>(
      p[0],
      p[1], p[2], p[3], p[4],   p[5], p[6], p[7], p[8],
      p[9], p[10], p[11], p[12], p[13], p[14], p[15], p[16],
      p[17], p[18], (float*)d_out);
}
